// LucidRainsMinimal_55611236549025
// MI455X (gfx1250) — matrix operand order repaired and checked by reading, not compiled, not run on hardware
//
#include <hip/hip_runtime.h>
#include <hip/hip_bf16.h>
#include <math.h>

typedef __attribute__((ext_vector_type(16))) _Float16 v16h;
typedef __attribute__((ext_vector_type(8)))  _Float16 v8h;
typedef __attribute__((ext_vector_type(8)))  float    v8f;
typedef _Float16 f16;

#define DIMX   512
#define HQ     8
#define KHN    2
#define GG     4
#define DHD    64
#define BALLW  128
#define SELB   64
#define PD     3
#define SEQL   4096
#define BATCHN 8
#define NTOKN  (BATCHN*SEQL)        // 32768
#define NBLK   (SEQL/SELB)          // 64
#define NWIN   (SEQL/BALLW)         // 32
#define CDIM   (SELB*DHD)           // 4096
#define SCL    0.125f               // 64^-0.5

// ---------------------------------------------------------------- helpers ---
__device__ __forceinline__ v8f zero8() {
  v8f z;
#pragma unroll
  for (int i = 0; i < 8; ++i) z[i] = 0.0f;
  return z;
}

__device__ __forceinline__ v8f wmma_f16(v16h a, v16h b, v8f c) {
  return __builtin_amdgcn_wmma_f32_16x16x32_f16(false, a, false, b, (short)0, c,
                                                false, false);
}

// A-matrix fragment (16x32 f16). Per CDNA5 ISA: lane half h, row m=lane%16.
// element e -> K = kBase + 8*h + ((e<8) ? e : e+8). Two contiguous 8-half runs.
__device__ __forceinline__ v16h fragA(const f16* base, int stride, int row,
                                      int kBase, int half) {
  const f16* p = base + row * stride + kBase + half * 8;
  v8h lo = *(const v8h*)(p);
  v8h hi = *(const v8h*)(p + 16);
  v16h r;
#pragma unroll
  for (int e = 0; e < 8; ++e) { r[e] = lo[e]; r[e + 8] = hi[e]; }
  return r;
}

// B-matrix fragment (32x16 f16). Per ISA B layout: lane half h, col n=lane%16,
// element e -> K = kBase + 8*h + ((e<8) ? e : e+8). Two contiguous 8-half runs.
__device__ __forceinline__ v16h fragB(const f16* base, int stride, int col,
                                      int kBase, int half) {
  const f16* p = base + col * stride + kBase + half * 8;
  v8h lo = *(const v8h*)(p);
  v8h hi = *(const v8h*)(p + 16);
  v16h r;
#pragma unroll
  for (int e = 0; e < 8; ++e) { r[e] = lo[e]; r[e + 8] = hi[e]; }
  return r;
}

// ---------------------------------------------------------------- misc ------
__global__ void zerof_kernel(float* p, int n) {
  int i = blockIdx.x * blockDim.x + threadIdx.x;
  if (i < n) p[i] = 0.0f;
}

__global__ void f32_to_f16_kernel(const float* __restrict__ s,
                                  f16* __restrict__ d, size_t n) {
  size_t i = (size_t)blockIdx.x * blockDim.x + threadIdx.x;
  if (i < n) d[i] = (f16)s[i];
}

// group means of pos over SEL-token groups
__global__ __launch_bounds__(64) void pre_mean_kernel(
    const float* __restrict__ pos, float* __restrict__ gmean) {
  int g = blockIdx.x;
  int j = threadIdx.x;
  __shared__ float s[PD][SELB];
  const float* p = pos + ((size_t)g * SELB + j) * PD;
  s[0][j] = p[0]; s[1][j] = p[1]; s[2][j] = p[2];
  __syncthreads();
  if (j < PD) {
    float acc = 0.0f;
    for (int t = 0; t < SELB; ++t) acc += s[j][t];
    gmean[g * PD + j] = acc * (1.0f / (float)SELB);
  }
}

// x += rel@pe_w + pe_b ; h = rmsnorm(x)*gamma  -> f16
__global__ __launch_bounds__(256) void pre_norm_kernel(
    const float* __restrict__ x, const float* __restrict__ pos,
    const float* __restrict__ gmean, const float* __restrict__ pe_w,
    const float* __restrict__ pe_b, const float* __restrict__ gamma,
    f16* __restrict__ h_out) {
  int t = blockIdx.x;
  int g = t / SELB;
  float r0 = pos[t * PD + 0] - gmean[g * PD + 0];
  float r1 = pos[t * PD + 1] - gmean[g * PD + 1];
  float r2 = pos[t * PD + 2] - gmean[g * PD + 2];
  __shared__ float red[8];
  float vals[2];
  float ss = 0.0f;
#pragma unroll
  for (int i = 0; i < 2; ++i) {
    int c = threadIdx.x * 2 + i;
    float v = x[(size_t)t * DIMX + c] + r0 * pe_w[0 * DIMX + c] +
              r1 * pe_w[1 * DIMX + c] + r2 * pe_w[2 * DIMX + c] + pe_b[c];
    vals[i] = v;
    ss += v * v;
  }
  for (int m = 16; m >= 1; m >>= 1) ss += __shfl_xor(ss, m, 32);
  if ((threadIdx.x & 31) == 0) red[threadIdx.x >> 5] = ss;
  __syncthreads();
  float tot = 0.0f;
#pragma unroll
  for (int w = 0; w < 8; ++w) tot += red[w];
  float rms = rsqrtf(tot / (float)DIMX + 1e-6f);
#pragma unroll
  for (int i = 0; i < 2; ++i) {
    int c = threadIdx.x * 2 + i;
    h_out[(size_t)t * DIMX + c] = (f16)(vals[i] * rms * gamma[c]);
  }
}

// GPT-NeoX rotary on (NTOK, nHeads*DHD) f32 -> f16
__global__ void rotary_kernel(const float* __restrict__ src,
                              f16* __restrict__ dst, int nHeads) {
  size_t idx = (size_t)blockIdx.x * blockDim.x + threadIdx.x;
  size_t total = (size_t)NTOKN * nHeads * (DHD / 2);
  if (idx >= total) return;
  int d = (int)(idx % (DHD / 2));
  size_t th = idx / (DHD / 2);
  int hh = (int)(th % nHeads);
  size_t t = th / nHeads;
  int s = (int)(t % SEQL);
  float inv = powf(10000.0f, -2.0f * (float)d / (float)DHD);
  float f = (float)s * inv;
  float c = cosf(f), sn = sinf(f);
  size_t base = t * (size_t)(nHeads * DHD) + (size_t)hh * DHD;
  float t1 = src[base + d], t2 = src[base + d + DHD / 2];
  dst[base + d] = (f16)(t1 * c - t2 * sn);
  dst[base + d + DHD / 2] = (f16)(t2 * c + t1 * sn);
}

// ---------------------------------------------------------------- GEMM ------
// C[M,N] = act(A[M,K] @ W[K,N] + bias). A,W f16 row-major. 256 thr, 8 waves.
template <int ACT, bool OUT32, bool OUT16>
__global__ __launch_bounds__(256) void gemm_wmma_kernel(
    const f16* __restrict__ A, const f16* __restrict__ W,
    const float* __restrict__ bias, float* __restrict__ C32,
    f16* __restrict__ C16, int M, int N, int K) {
  constexpr int BM = 64, BN = 128, BK = 32, AST = 40, BST = 40;
  __shared__ __attribute__((aligned(16))) f16 As[BM * AST];
  __shared__ __attribute__((aligned(16))) f16 Bs[BN * BST];  // Bs[n][k]
  int tid = threadIdx.x;
  int lane = tid & 31, wave = tid >> 5;
  int half = lane >> 4, l16 = lane & 15;
  int m0 = blockIdx.x * BM, n0 = blockIdx.y * BN;
  int waveM = wave >> 2, waveN = wave & 3;  // 2 x 4 waves
  v8f acc[2][2];
#pragma unroll
  for (int a = 0; a < 2; ++a)
#pragma unroll
    for (int b = 0; b < 2; ++b) acc[a][b] = zero8();

  int ar = tid >> 2;             // A row within tile
  int ac = (tid & 3) * 8;        // A col chunk
  int bn = tid & 127;            // B col
  int bk0 = tid >> 7;            // 0..1

  for (int k0 = 0; k0 < K; k0 += BK) {
    {  // A tile
      v8h val;
#pragma unroll
      for (int e = 0; e < 8; ++e) val[e] = (f16)0;
      if (m0 + ar < M) val = *(const v8h*)(A + (size_t)(m0 + ar) * K + k0 + ac);
      *(v8h*)&As[ar * AST + ac] = val;
    }
    {  // B tile, transposed into Bs[n][k]
      if (k0 + BK < K)
        __builtin_prefetch(W + (size_t)(k0 + BK + bk0) * N + n0 + bn, 0, 1);
#pragma unroll
      for (int kk = 0; kk < BK; kk += 2) {
        int k = kk + bk0;
        f16 v = (f16)0;
        if (n0 + bn < N) v = W[(size_t)(k0 + k) * N + n0 + bn];
        Bs[bn * BST + k] = v;
      }
    }
    __syncthreads();
#pragma unroll
    for (int mi = 0; mi < 2; ++mi) {
      v16h a = fragA(As, AST, waveM * 32 + mi * 16 + l16, 0, half);
#pragma unroll
      for (int ni = 0; ni < 2; ++ni) {
        v16h b = fragB(Bs, BST, waveN * 32 + ni * 16 + l16, 0, half);
        acc[mi][ni] = wmma_f16(a, b, acc[mi][ni]);
      }
    }
    __syncthreads();
  }
#pragma unroll
  for (int mi = 0; mi < 2; ++mi)
#pragma unroll
    for (int ni = 0; ni < 2; ++ni) {
      int n = n0 + waveN * 32 + ni * 16 + l16;
      if (n >= N) continue;
      float bv = bias ? bias[n] : 0.0f;
#pragma unroll
      for (int r = 0; r < 8; ++r) {
        int m = m0 + waveM * 32 + mi * 16 + r + 8 * half;
        if (m >= M) continue;
        float v = acc[mi][ni][r] + bv;
        if (ACT == 1) v = fmaxf(v, 0.0f);
        if (ACT == 2) v = 1.0f / (1.0f + expf(-v));
        if (OUT32) C32[(size_t)m * N + n] = v;
        if (OUT16) C16[(size_t)m * N + n] = (f16)v;
      }
    }
}

// ------------------------------------------------------- compression prep ---
// cin[(b*KH+kh)*NB+n][s*DHD+d] = kv32[token][kh*DHD+d] + posemb[kh][s][d]
__global__ __launch_bounds__(256) void build_cin_kernel(
    const float* __restrict__ kv32, const float* __restrict__ posemb,
    f16* __restrict__ cin) {
  int row = blockIdx.x;
  int n = row % NBLK;
  int kh = (row / NBLK) % KHN;
  int b = row / (NBLK * KHN);
  for (int i = threadIdx.x; i < CDIM; i += 256) {
    int s = i / DHD, d = i % DHD;
    size_t tok = (size_t)b * SEQL + (size_t)n * SELB + s;
    float v = kv32[tok * (KHN * DHD) + kh * DHD + d] +
              posemb[((size_t)kh * SELB + s) * DHD + d];
    cin[(size_t)row * CDIM + i] = (f16)v;
  }
}

// pack ck/cv: row0 = mem, rows 1..64 = mlp output, rows 65..95 = 0. f16.
__global__ void pack_ck_kernel(const float* __restrict__ core,
                               const float* __restrict__ mem,
                               f16* __restrict__ out) {
  int idx = blockIdx.x * blockDim.x + threadIdx.x;
  const int total = BATCHN * KHN * 96 * DHD;
  if (idx >= total) return;
  int d = idx % DHD;
  int r = (idx / DHD) % 96;
  int kh = (idx / (DHD * 96)) % KHN;
  int b = idx / (DHD * 96 * KHN);
  float v = 0.0f;
  if (r == 0) v = mem[kh * DHD + d];
  else if (r <= NBLK)
    v = core[(size_t)((b * KHN + kh) * NBLK + (r - 1)) * DHD + d];
  out[idx] = (f16)v;
}

// ---------------------------------------------------- compressed attention --
// block per (b,kh,g,qblock of 64). 128 thr = 4 waves x 16 q rows.
__global__ __launch_bounds__(128) void attn_compressed_kernel(
    const f16* __restrict__ qr, const f16* __restrict__ ck,
    const f16* __restrict__ cv, f16* __restrict__ outb,
    float* __restrict__ imp) {
  constexpr int KEYS = 96, QST = 72, CKST = 72, VST = 104, PST = 104;
  __shared__ __attribute__((aligned(16))) f16 qs[64 * QST];
  __shared__ __attribute__((aligned(16))) f16 cks[KEYS * CKST];
  __shared__ __attribute__((aligned(16))) f16 cvt[DHD * VST];
  __shared__ __attribute__((aligned(16))) f16 ps[64 * PST];
  __shared__ float impl[NBLK];
  int blk = blockIdx.x;
  int qb = blk % NBLK;
  int g = (blk / NBLK) % GG;
  int kh = (blk / (NBLK * GG)) % KHN;
  int b = blk / (NBLK * GG * KHN);
  int hh = kh * GG + g;
  int tid = threadIdx.x, lane = tid & 31, wave = tid >> 5;
  int half = lane >> 4, l16 = lane & 15;
  size_t tok0 = (size_t)b * SEQL + (size_t)qb * SELB;

  for (int i = tid; i < 64 * DHD / 8; i += 128) {
    int r = i >> 3, c8 = (i & 7) * 8;
    *(v8h*)&qs[r * QST + c8] =
        *(const v8h*)&qr[(tok0 + r) * (HQ * DHD) + hh * DHD + c8];
  }
  const f16* ckg = ck + (size_t)(b * KHN + kh) * KEYS * DHD;
  for (int i = tid; i < KEYS * DHD / 8; i += 128) {
    int r = i >> 3, c8 = (i & 7) * 8;
    *(v8h*)&cks[r * CKST + c8] = *(const v8h*)&ckg[r * DHD + c8];
  }
  const f16* cvg = cv + (size_t)(b * KHN + kh) * KEYS * DHD;
  for (int i = tid; i < KEYS * DHD; i += 128) {
    int key = i / DHD, d = i % DHD;
    cvt[d * VST + key] = cvg[key * DHD + d];
  }
  for (int i = tid; i < NBLK; i += 128) impl[i] = 0.0f;
  __syncthreads();

  int qBase = wave * 16;
  v8f sc[6];
#pragma unroll
  for (int tn = 0; tn < 6; ++tn) {
    v8f a = zero8();
#pragma unroll
    for (int k0 = 0; k0 < DHD; k0 += 32)
      a = wmma_f16(fragA(qs, QST, qBase + l16, k0, half),
                   fragB(cks, CKST, tn * 16 + l16, k0, half), a);
    sc[tn] = a;
  }
#pragma unroll
  for (int r = 0; r < 8; ++r) {
    float mx = -1e30f;
#pragma unroll
    for (int tn = 0; tn < 6; ++tn) {
      int key = tn * 16 + l16;
      float v = sc[tn][r] * SCL;
      sc[tn][r] = v;
      if (key < 65) mx = fmaxf(mx, v);
    }
    for (int m = 8; m >= 1; m >>= 1) mx = fmaxf(mx, __shfl_xor(mx, m, 32));
    float pv[6];
    float sum = 0.0f;
#pragma unroll
    for (int tn = 0; tn < 6; ++tn) {
      int key = tn * 16 + l16;
      float p = (key < 65) ? expf(sc[tn][r] - mx) : 0.0f;
      pv[tn] = p;
      sum += p;
    }
    for (int m = 8; m >= 1; m >>= 1) sum += __shfl_xor(sum, m, 32);
    float inv = 1.0f / sum;
    int rowm = qBase + r + 8 * half;
#pragma unroll
    for (int tn = 0; tn < 6; ++tn) {
      float p = pv[tn] * inv;
      ps[rowm * PST + tn * 16 + l16] = (f16)p;
      int key = tn * 16 + l16;
      if (key >= 1 && key <= NBLK) atomicAdd(&impl[key - 1], p);
    }
  }
  __syncthreads();
#pragma unroll
  for (int dn = 0; dn < 4; ++dn) {
    v8f a = zero8();
#pragma unroll
    for (int k0 = 0; k0 < KEYS; k0 += 32)
      a = wmma_f16(fragA(ps, PST, qBase + l16, k0, half),
                   fragB(cvt, VST, dn * 16 + l16, k0, half), a);
#pragma unroll
    for (int r = 0; r < 8; ++r) {
      int rowm = qBase + r + 8 * half;
      outb[(tok0 + rowm) * (HQ * DHD) + hh * DHD + dn * 16 + l16] = (f16)a[r];
    }
  }
  __syncthreads();
  float* impg = imp + (size_t)((b * KHN + kh) * NBLK + qb) * NBLK;
  for (int i = tid; i < NBLK; i += 128) atomicAdd(&impg[i], impl[i]);
}

__global__ void topk2_kernel(const float* __restrict__ imp,
                             int* __restrict__ sel) {
  int row = blockIdx.x * blockDim.x + threadIdx.x;
  if (row >= BATCHN * KHN * NBLK) return;
  const float* p = imp + (size_t)row * NBLK;
  float b0 = -1e30f, b1 = -1e30f;
  int i0 = 0, i1 = 0;
  for (int j = 0; j < NBLK; ++j) {
    float v = p[j];
    if (v > b0) { b1 = b0; i1 = i0; b0 = v; i0 = j; }
    else if (v > b1) { b1 = v; i1 = j; }
  }
  sel[row * 2 + 0] = i0;
  sel[row * 2 + 1] = i1;
}

// -------------------------------- flash attention (sliding + fine branch) ---
// mode 0: sliding ball (2 key chunks of 64 within the 128-window).
// mode 1: fine (chunks = sel0, sel1, own block).
// 128 thr = 4 waves x 16 q rows (64 q rows / block). Online softmax.
__global__ __launch_bounds__(128) void attn_flash_kernel(
    const f16* __restrict__ qr, const f16* __restrict__ kr,
    const f16* __restrict__ vh, const int* __restrict__ selb,
    f16* __restrict__ outb, int mode) {
  constexpr int ST = 72;
  __shared__ __attribute__((aligned(16))) f16 qs[64 * ST];
  __shared__ __attribute__((aligned(16))) f16 ks[64 * ST];
  __shared__ __attribute__((aligned(16))) f16 vt[64 * ST];
  __shared__ __attribute__((aligned(16))) f16 ps[64 * ST];
  int blk = blockIdx.x;
  int b, hh, kh, nchunks;
  size_t qtok0;
  size_t ctok[3];
  if (mode == 0) {
    int qhalf = blk & 1;
    int t = blk >> 1;
    int w = t % NWIN;
    hh = (t / NWIN) % HQ;
    b = t / (NWIN * HQ);
    kh = hh / GG;
    nchunks = 2;
    size_t wtok = (size_t)b * SEQL + (size_t)w * BALLW;
    ctok[0] = wtok;
    ctok[1] = wtok + 64;
    ctok[2] = wtok;
    qtok0 = wtok + (size_t)qhalf * 64;
  } else {
    int qb = blk % NBLK;
    int g = (blk / NBLK) % GG;
    kh = (blk / (NBLK * GG)) % KHN;
    b = blk / (NBLK * GG * KHN);
    hh = kh * GG + g;
    nchunks = 3;
    const int* sp = selb + (size_t)((b * KHN + kh) * NBLK + qb) * 2;
    ctok[0] = (size_t)b * SEQL + (size_t)sp[0] * SELB;
    ctok[1] = (size_t)b * SEQL + (size_t)sp[1] * SELB;
    ctok[2] = (size_t)b * SEQL + (size_t)qb * SELB;
    qtok0 = (size_t)b * SEQL + (size_t)qb * SELB;
  }
  int tid = threadIdx.x, lane = tid & 31, wave = tid >> 5;
  int half = lane >> 4, l16 = lane & 15;
  for (int i = tid; i < 64 * DHD / 8; i += 128) {
    int r = i >> 3, c8 = (i & 7) * 8;
    *(v8h*)&qs[r * ST + c8] =
        *(const v8h*)&qr[(qtok0 + r) * (HQ * DHD) + hh * DHD + c8];
  }
  int qBase = wave * 16;
  v8f acc[4];
#pragma unroll
  for (int dn = 0; dn < 4; ++dn) acc[dn] = zero8();
  float mrun[8], lrun[8];
#pragma unroll
  for (int r = 0; r < 8; ++r) { mrun[r] = -1e30f; lrun[r] = 0.0f; }

  for (int c = 0; c < nchunks; ++c) {
    __syncthreads();
    for (int i = tid; i < 64 * DHD / 8; i += 128) {
      int r = i >> 3, c8 = (i & 7) * 8;
      *(v8h*)&ks[r * ST + c8] =
          *(const v8h*)&kr[(ctok[c] + r) * (KHN * DHD) + kh * DHD + c8];
    }
    for (int i = tid; i < 64 * DHD; i += 128) {
      int key = i >> 6, d = i & 63;
      vt[d * ST + key] = vh[(ctok[c] + key) * (KHN * DHD) + kh * DHD + d];
    }
    __syncthreads();
    v8f sc[4];
#pragma unroll
    for (int tn = 0; tn < 4; ++tn) {
      v8f a = zero8();
#pragma unroll
      for (int k0 = 0; k0 < DHD; k0 += 32)
        a = wmma_f16(fragA(qs, ST, qBase + l16, k0, half),
                     fragB(ks, ST, tn * 16 + l16, k0, half), a);
      sc[tn] = a;
    }
    float fac[8];
#pragma unroll
    for (int r = 0; r < 8; ++r) {
      float mc = -1e30f;
#pragma unroll
      for (int tn = 0; tn < 4; ++tn) {
        float v = sc[tn][r] * SCL;
        sc[tn][r] = v;
        mc = fmaxf(mc, v);
      }
      for (int m = 8; m >= 1; m >>= 1) mc = fmaxf(mc, __shfl_xor(mc, m, 32));
      float mnew = fmaxf(mrun[r], mc);
      float f = expf(mrun[r] - mnew);
      float rs = 0.0f;
      int rowm = qBase + r + 8 * half;
#pragma unroll
      for (int tn = 0; tn < 4; ++tn) {
        float p = expf(sc[tn][r] - mnew);
        ps[rowm * ST + tn * 16 + l16] = (f16)p;
        rs += p;
      }
      for (int m = 8; m >= 1; m >>= 1) rs += __shfl_xor(rs, m, 32);
      lrun[r] = lrun[r] * f + rs;
      mrun[r] = mnew;
      fac[r] = f;
    }
#pragma unroll
    for (int dn = 0; dn < 4; ++dn) {
#pragma unroll
      for (int r = 0; r < 8; ++r) acc[dn][r] *= fac[r];
#pragma unroll
      for (int k0 = 0; k0 < 64; k0 += 32)
        acc[dn] = wmma_f16(fragA(ps, ST, qBase + l16, k0, half),
                           fragB(vt, ST, dn * 16 + l16, k0, half), acc[dn]);
    }
  }
#pragma unroll
  for (int dn = 0; dn < 4; ++dn)
#pragma unroll
    for (int r = 0; r < 8; ++r) {
      int rowm = qBase + r + 8 * half;
      outb[(qtok0 + rowm) * (HQ * DHD) + hh * DHD + dn * 16 + l16] =
          (f16)(acc[dn][r] / lrun[r]);
    }
}

// ---------------------------------------------------------------- combine ---
__global__ void combine_kernel(const float* __restrict__ gates,
                               const f16* __restrict__ c,
                               const f16* __restrict__ f,
                               const f16* __restrict__ s,
                               f16* __restrict__ out) {
  size_t idx = (size_t)blockIdx.x * blockDim.x + threadIdx.x;
  if (idx >= (size_t)NTOKN * DIMX) return;
  size_t t = idx / DIMX;
  int col = (int)(idx % DIMX);
  int hh = col / DHD;
  const float* gp = gates + t * (3 * HQ) + hh * 3;
  float v = gp[0] * (float)c[idx] + gp[1] * (float)f[idx] + gp[2] * (float)s[idx];
  out[idx] = (f16)v;
}

// ---------------------------------------------------------------- launch ----
extern "C" void kernel_launch(void* const* d_in, const int* in_sizes, int n_in,
                              void* d_out, int out_size, void* d_ws,
                              size_t ws_size, hipStream_t stream) {
  (void)in_sizes; (void)n_in; (void)out_size;
  const float* x     = (const float*)d_in[0];
  const float* pos   = (const float*)d_in[1];
  const float* pe_w  = (const float*)d_in[2];
  const float* pe_b  = (const float*)d_in[3];
  const float* gamma = (const float*)d_in[4];
  const float* wq    = (const float*)d_in[5];
  const float* wk    = (const float*)d_in[6];
  const float* wv    = (const float*)d_in[7];
  const float* k_pos = (const float*)d_in[8];
  const float* v_pos = (const float*)d_in[9];
  const float* k_w1  = (const float*)d_in[10];
  const float* k_b1  = (const float*)d_in[11];
  const float* k_w2  = (const float*)d_in[12];
  const float* k_b2  = (const float*)d_in[13];
  const float* v_w1  = (const float*)d_in[14];
  const float* v_b1  = (const float*)d_in[15];
  const float* v_w2  = (const float*)d_in[16];
  const float* v_b2  = (const float*)d_in[17];
  const float* mem_k = (const float*)d_in[18];
  const float* mem_v = (const float*)d_in[19];
  const float* w_cmb = (const float*)d_in[20];
  const float* w_out = (const float*)d_in[21];
  float* outp = (float*)d_out;

  char* base = (char*)d_ws;
  size_t off = 0;
  auto alloc = [&](size_t bytes) -> void* {
    void* r = base + off;
    off = (off + bytes + 255) & ~(size_t)255;
    return r;
  };
  const size_t NT = NTOKN;
  f16* h_h     = (f16*)alloc(NT * DIMX * 2);
  f16* wq_h    = (f16*)alloc((size_t)DIMX * DIMX * 2);
  f16* wk_h    = (f16*)alloc((size_t)DIMX * KHN * DHD * 2);
  f16* wv_h    = (f16*)alloc((size_t)DIMX * KHN * DHD * 2);
  f16* wcmb_h  = (f16*)alloc((size_t)DIMX * 3 * HQ * 2);
  f16* wout_h  = (f16*)alloc((size_t)DIMX * DIMX * 2);
  f16* kw1_h   = (f16*)alloc((size_t)CDIM * CDIM * 2);
  f16* kw2_h   = (f16*)alloc((size_t)CDIM * DHD * 2);
  f16* vw1_h   = (f16*)alloc((size_t)CDIM * CDIM * 2);
  f16* vw2_h   = (f16*)alloc((size_t)CDIM * DHD * 2);
  float* q32   = (float*)alloc(NT * DIMX * 4);      // reused for cout+fout
  float* k32   = (float*)alloc(NT * KHN * DHD * 4);
  float* v32   = (float*)alloc(NT * KHN * DHD * 4);
  f16* qr_h    = (f16*)alloc(NT * DIMX * 2);
  f16* kr_h    = (f16*)alloc(NT * KHN * DHD * 2);
  f16* v_h     = (f16*)alloc(NT * KHN * DHD * 2);
  float* gates = (float*)alloc(NT * 3 * HQ * 4);
  float* gmean = (float*)alloc((NT / SELB) * PD * 4);
  f16* cin_k   = (f16*)alloc((size_t)BATCHN * KHN * NBLK * CDIM * 2);
  f16* cin_v   = (f16*)alloc((size_t)BATCHN * KHN * NBLK * CDIM * 2);
  f16* hid     = (f16*)alloc((size_t)BATCHN * KHN * NBLK * CDIM * 2);
  float* ckc   = (float*)alloc((size_t)BATCHN * KHN * NBLK * DHD * 4);
  float* cvc   = (float*)alloc((size_t)BATCHN * KHN * NBLK * DHD * 4);
  f16* ck_h    = (f16*)alloc((size_t)BATCHN * KHN * 96 * DHD * 2);
  f16* cv_h    = (f16*)alloc((size_t)BATCHN * KHN * 96 * DHD * 2);
  float* imp   = (float*)alloc((size_t)BATCHN * KHN * NBLK * NBLK * 4);
  int* selbuf  = (int*)alloc((size_t)BATCHN * KHN * NBLK * 2 * 4);
  f16* sout    = (f16*)alloc(NT * DIMX * 2);
  f16* comb    = (f16*)alloc(NT * DIMX * 2);
  if (off > ws_size) return;  // workspace too small: bail deterministically
  f16* cout = (f16*)q32;                 // alias (q32 dead after rotary)
  f16* fout = cout + NT * DIMX;

  auto cvt = [&](const float* s, f16* d, size_t n) {
    f32_to_f16_kernel<<<(unsigned)((n + 255) / 256), 256, 0, stream>>>(s, d, n);
  };
  // weights -> f16
  cvt(wq, wq_h, (size_t)DIMX * DIMX);
  cvt(wk, wk_h, (size_t)DIMX * KHN * DHD);
  cvt(wv, wv_h, (size_t)DIMX * KHN * DHD);
  cvt(w_cmb, wcmb_h, (size_t)DIMX * 3 * HQ);
  cvt(w_out, wout_h, (size_t)DIMX * DIMX);
  cvt(k_w1, kw1_h, (size_t)CDIM * CDIM);
  cvt(k_w2, kw2_h, (size_t)CDIM * DHD);
  cvt(v_w1, vw1_h, (size_t)CDIM * CDIM);
  cvt(v_w2, vw2_h, (size_t)CDIM * DHD);

  // preprocess + rmsnorm
  pre_mean_kernel<<<NT / SELB, 64, 0, stream>>>(pos, gmean);
  pre_norm_kernel<<<(unsigned)NT, 256, 0, stream>>>(x, pos, gmean, pe_w, pe_b,
                                                    gamma, h_h);
  // projections
  gemm_wmma_kernel<0, true, false><<<dim3(NT / 64, DIMX / 128), 256, 0, stream>>>(
      h_h, wq_h, (const float*)nullptr, q32, (f16*)nullptr, (int)NT, DIMX, DIMX);
  gemm_wmma_kernel<0, true, false><<<dim3(NT / 64, 1), 256, 0, stream>>>(
      h_h, wk_h, (const float*)nullptr, k32, (f16*)nullptr, (int)NT, KHN * DHD, DIMX);
  gemm_wmma_kernel<0, true, false><<<dim3(NT / 64, 1), 256, 0, stream>>>(
      h_h, wv_h, (const float*)nullptr, v32, (f16*)nullptr, (int)NT, KHN * DHD, DIMX);
  gemm_wmma_kernel<2, true, false><<<dim3(NT / 64, 1), 256, 0, stream>>>(
      h_h, wcmb_h, (const float*)nullptr, gates, (f16*)nullptr, (int)NT, 3 * HQ, DIMX);
  // rotary + v to f16
  rotary_kernel<<<(unsigned)((NT * HQ * (DHD / 2) + 255) / 256), 256, 0, stream>>>(
      q32, qr_h, HQ);
  rotary_kernel<<<(unsigned)((NT * KHN * (DHD / 2) + 255) / 256), 256, 0, stream>>>(
      k32, kr_h, KHN);
  cvt(v32, v_h, NT * KHN * DHD);
  // compression MLPs
  const int MR = BATCHN * KHN * NBLK;  // 1024
  build_cin_kernel<<<MR, 256, 0, stream>>>(k32, k_pos, cin_k);
  build_cin_kernel<<<MR, 256, 0, stream>>>(v32, v_pos, cin_v);
  gemm_wmma_kernel<1, false, true><<<dim3(MR / 64, CDIM / 128), 256, 0, stream>>>(
      cin_k, kw1_h, k_b1, (float*)nullptr, hid, MR, CDIM, CDIM);
  gemm_wmma_kernel<0, true, false><<<dim3(MR / 64, 1), 256, 0, stream>>>(
      hid, kw2_h, k_b2, ckc, (f16*)nullptr, MR, DHD, CDIM);
  gemm_wmma_kernel<1, false, true><<<dim3(MR / 64, CDIM / 128), 256, 0, stream>>>(
      cin_v, vw1_h, v_b1, (float*)nullptr, hid, MR, CDIM, CDIM);
  gemm_wmma_kernel<0, true, false><<<dim3(MR / 64, 1), 256, 0, stream>>>(
      hid, vw2_h, v_b2, cvc, (f16*)nullptr, MR, DHD, CDIM);
  pack_ck_kernel<<<(BATCHN * KHN * 96 * DHD + 255) / 256, 256, 0, stream>>>(
      ckc, mem_k, ck_h);
  pack_ck_kernel<<<(BATCHN * KHN * 96 * DHD + 255) / 256, 256, 0, stream>>>(
      cvc, mem_v, cv_h);
  // compressed attention + importance + top-k
  zerof_kernel<<<(BATCHN * KHN * NBLK * NBLK + 255) / 256, 256, 0, stream>>>(
      imp, BATCHN * KHN * NBLK * NBLK);
  attn_compressed_kernel<<<BATCHN * KHN * GG * NBLK, 128, 0, stream>>>(
      qr_h, ck_h, cv_h, cout, imp);
  topk2_kernel<<<(BATCHN * KHN * NBLK + 127) / 128, 128, 0, stream>>>(imp, selbuf);
  // fine + sliding branches (flash)
  attn_flash_kernel<<<BATCHN * KHN * GG * NBLK, 128, 0, stream>>>(
      qr_h, kr_h, v_h, selbuf, fout, 1);
  attn_flash_kernel<<<BATCHN * HQ * NWIN * 2, 128, 0, stream>>>(
      qr_h, kr_h, v_h, selbuf, sout, 0);
  // gated combine + output projection
  combine_kernel<<<(unsigned)((NT * DIMX + 255) / 256), 256, 0, stream>>>(
      gates, cout, fout, sout, comb);
  gemm_wmma_kernel<0, true, false><<<dim3(NT / 64, DIMX / 128), 256, 0, stream>>>(
      comb, wout_h, (const float*)nullptr, outp, (f16*)nullptr, (int)NT, DIMX, DIMX);
}
